// MixtureOfExperts_24309514895718
// MI455X (gfx1250) — matrix operand order repaired and checked by reading, not compiled, not run on hardware
//
#include <hip/hip_runtime.h>
#include <math.h>

#define D_MODEL 1024
#define FFN_H   4096
#define N_EXP   8
#define N_TOK   8192
#define CAP     8192       // max tokens routed to one expert (<= T)
#define BM      32         // token tile per block
#define HC      64         // FFN-hidden chunk

typedef __attribute__((ext_vector_type(16))) __bf16 v16bf;
typedef __attribute__((ext_vector_type(8)))  __bf16 v8bf;
typedef __attribute__((ext_vector_type(4)))  __bf16 v4bf;
typedef __attribute__((ext_vector_type(2)))  __bf16 v2bf;
typedef __attribute__((ext_vector_type(8)))  float  v8f;

// Build a 16-element bf16 WMMA fragment from two contiguous 16-byte LDS chunks.
__device__ __forceinline__ v16bf frag16(const __bf16* lo, const __bf16* hi) {
  v8bf a = *(const v8bf*)lo;
  v8bf b = *(const v8bf*)hi;
  v16bf r;
#pragma unroll
  for (int i = 0; i < 8; ++i) { r[i] = a[i]; r[i + 8] = b[i]; }
  return r;
}

__device__ __forceinline__ v2bf pack2(float a, float b) {
  v2bf p; p[0] = (__bf16)a; p[1] = (__bf16)b; return p;  // -> v_cvt_pk_bf16_f32
}

// ---------------------------------------------------------------- zero pass
__global__ __launch_bounds__(256) void moe_zero_kernel(float* __restrict__ out,
                                                       int* __restrict__ cnt) {
  int i = blockIdx.x * 256 + threadIdx.x;
  if (i < N_TOK * D_MODEL) out[i] = 0.0f;
  if (i < N_EXP) cnt[i] = 0;
}

// ---------------------------------------------------------------- gating
__global__ __launch_bounds__(256) void moe_gate_kernel(
    const float* __restrict__ x, const float* __restrict__ wg,
    int* __restrict__ cnt, int* __restrict__ tok, float* __restrict__ wgt) {
  __shared__ float sg[N_EXP * D_MODEL];            // 32 KB
  for (int i = threadIdx.x; i < N_EXP * D_MODEL; i += 256) sg[i] = wg[i];
  __syncthreads();

  int t = blockIdx.x * 256 + threadIdx.x;
  const float* xr = x + (size_t)t * D_MODEL;
  float l[N_EXP];
#pragma unroll
  for (int e = 0; e < N_EXP; ++e) l[e] = 0.0f;
  for (int d = 0; d < D_MODEL; d += 4) {
    float4 xv = *(const float4*)(xr + d);
#pragma unroll
    for (int e = 0; e < N_EXP; ++e) {
      const float* g = sg + e * D_MODEL + d;
      l[e] += xv.x * g[0] + xv.y * g[1] + xv.z * g[2] + xv.w * g[3];
    }
  }
  // softmax over 8
  float mx = l[0];
#pragma unroll
  for (int e = 1; e < N_EXP; ++e) mx = fmaxf(mx, l[e]);
  float s = 0.0f, p[N_EXP];
#pragma unroll
  for (int e = 0; e < N_EXP; ++e) { p[e] = __expf(l[e] - mx); s += p[e]; }
  float inv = 1.0f / s;
#pragma unroll
  for (int e = 0; e < N_EXP; ++e) p[e] *= inv;
  // top-2 (strict > keeps lowest index on ties, matching jax top_k)
  int i1 = 0; float g1 = p[0];
#pragma unroll
  for (int e = 1; e < N_EXP; ++e) if (p[e] > g1) { g1 = p[e]; i1 = e; }
  int i2 = (i1 == 0) ? 1 : 0; float g2 = p[i2];
#pragma unroll
  for (int e = 0; e < N_EXP; ++e)
    if (e != i1 && p[e] > g2) { g2 = p[e]; i2 = e; }

  int s1 = atomicAdd(&cnt[i1], 1);
  tok[i1 * CAP + s1] = t; wgt[i1 * CAP + s1] = g1;
  int s2 = atomicAdd(&cnt[i2], 1);
  tok[i2 * CAP + s2] = t; wgt[i2 * CAP + s2] = g2;
}

// ---------------------------------------------------------------- expert FFN
__global__ __launch_bounds__(256) void moe_ffn_kernel(
    const float* __restrict__ x,
    const float* __restrict__ W1, const float* __restrict__ b1,
    const float* __restrict__ W2, const float* __restrict__ b2,
    const int* __restrict__ cnt, const int* __restrict__ tok,
    const float* __restrict__ wgt, float* __restrict__ out) {
  const int e    = blockIdx.y;
  const int tile = blockIdx.x;
  const int nTok = cnt[e];
  if (tile * BM >= nTok) return;

  __shared__ __align__(16) __bf16 x_lds[BM][D_MODEL];    // 64 KB, [m][k]
  __shared__ __align__(16) __bf16 w1t[2][HC][BM];        // 8 KB,  double-buffered [n][k]
  __shared__ __align__(16) __bf16 h_lds[BM][HC];         // 4 KB,  [m][k]
  __shared__ __align__(16) __bf16 w2t[8][2][16][BM];     // 16 KB, per-wave double buf [n][k]
  __shared__ int   s_tok[BM];
  __shared__ float s_wgt[BM];

  const int tid  = threadIdx.x;
  const int lane = tid & 31;
  const int w    = tid >> 5;          // wave id 0..7
  const int ln16 = lane & 15;
  const int lhi  = lane >> 4;         // 0 or 1

  if (tid < BM) {
    int slot = tile * BM + tid;
    int tk = 0; float wt = 0.0f;
    if (slot < nTok) { tk = tok[e * CAP + slot]; wt = wgt[e * CAP + slot]; }
    s_tok[tid] = tk; s_wgt[tid] = wt;
  }
  __syncthreads();

  // stage x tile -> bf16 LDS; packed 4x bf16 stores (ds_store_b64)
  {
    int m  = tid >> 3;
    int c0 = (tid & 7) * 128;
    const float* xr = x + (size_t)s_tok[m] * D_MODEL + c0;
#pragma unroll 4
    for (int i = 0; i < 128; i += 4) {
      float4 v = *(const float4*)(xr + i);
      v4bf q;
      q[0] = (__bf16)v.x; q[1] = (__bf16)v.y;
      q[2] = (__bf16)v.z; q[3] = (__bf16)v.w;
      *(v4bf*)&x_lds[m][c0 + i] = q;
    }
  }

  const float* W1e = W1 + (size_t)e * D_MODEL * FFN_H;
  const float* W2e = W2 + (size_t)e * FFN_H * D_MODEL;

  const int mi = w & 1;               // stage-A M tile (0..1)
  const int ni = w >> 1;              // stage-A N tile (0..3)
  const int kp1 = (tid >> 4) * 2;     // W1 staging: k-pair base 0..30
  const int n1  = (tid & 15) * 4;     // W1 staging: n col 0..60

  v8f accy[2][8];
  {
    v8f z = {};
#pragma unroll
    for (int a = 0; a < 2; ++a)
#pragma unroll
      for (int b = 0; b < 8; ++b) accy[a][b] = z;
  }

  // W1 slab store: transposed [n][k], packed k-pairs (ds_store_b32)
  auto stw1 = [&](int buf, float4 va, float4 vb) {  // va=row kp1, vb=row kp1+1
    *(v2bf*)&w1t[buf][n1 + 0][kp1] = pack2(va.x, vb.x);
    *(v2bf*)&w1t[buf][n1 + 1][kp1] = pack2(va.y, vb.y);
    *(v2bf*)&w1t[buf][n1 + 2][kp1] = pack2(va.z, vb.z);
    *(v2bf*)&w1t[buf][n1 + 3][kp1] = pack2(va.w, vb.w);
  };
  // W2 slab helpers (per-wave private region): k-pair packed stores
  auto ldw2 = [&](int s, int h0, float4* v) {
    int kk2  = (s >> 3) * 32;
    int ncol = w * 128 + (s & 7) * 16;
    const float* base = W2e + (size_t)(h0 + kk2 + (lane >> 2) * 2) * D_MODEL +
                        ncol + (lane & 3) * 4;
    v[0] = *(const float4*)(base);
    v[1] = *(const float4*)(base + D_MODEL);
    v[2] = *(const float4*)(base + (size_t)16 * D_MODEL);
    v[3] = *(const float4*)(base + (size_t)17 * D_MODEL);
  };
  auto stw2 = [&](int buf, const float4* v) {
    int kp  = (lane >> 2) * 2;
    int nc4 = (lane & 3) * 4;
    *(v2bf*)&w2t[w][buf][nc4 + 0][kp]      = pack2(v[0].x, v[1].x);
    *(v2bf*)&w2t[w][buf][nc4 + 1][kp]      = pack2(v[0].y, v[1].y);
    *(v2bf*)&w2t[w][buf][nc4 + 2][kp]      = pack2(v[0].z, v[1].z);
    *(v2bf*)&w2t[w][buf][nc4 + 3][kp]      = pack2(v[0].w, v[1].w);
    *(v2bf*)&w2t[w][buf][nc4 + 0][kp + 16] = pack2(v[2].x, v[3].x);
    *(v2bf*)&w2t[w][buf][nc4 + 1][kp + 16] = pack2(v[2].y, v[3].y);
    *(v2bf*)&w2t[w][buf][nc4 + 2][kp + 16] = pack2(v[2].z, v[3].z);
    *(v2bf*)&w2t[w][buf][nc4 + 3][kp + 16] = pack2(v[2].w, v[3].w);
  };

  for (int h0 = 0; h0 < FFN_H; h0 += HC) {
    // ---------------- stage A: h = gelu(x @ W1[:, h0:h0+64] + b1)
    v8f acch = {};
    {  // prologue: stage slab kk=0 into buf 0
      const float* p = W1e + (size_t)kp1 * FFN_H + h0 + n1;
      float4 va = *(const float4*)p;
      float4 vb = *(const float4*)(p + FFN_H);
      stw1(0, va, vb);
    }
    for (int i = 0; i < 32; ++i) {
      int kk = i * 32;
      __syncthreads();   // slab i visible to all (compiler waits dscnt pre-barrier)
      // fragments from current buffer first (no pending stores ahead of them)
      int ml = mi * 16 + ln16;
      int ks = lhi * 8;
      v16bf a = frag16(&x_lds[ml][kk + ks], &x_lds[ml][kk + ks + 16]);
      int nl = ni * 16 + ln16;
      v16bf b = frag16(&w1t[i & 1][nl][lhi * 8], &w1t[i & 1][nl][lhi * 8 + 16]);
      // issue next slab's global loads while WMMA runs
      float4 va, vb;
      bool hasNext = (i + 1 < 32);
      if (hasNext) {
        const float* p = W1e + (size_t)(kk + 32 + kp1) * FFN_H + h0 + n1;
        if (i + 2 < 32)
          __builtin_prefetch(p + 32 * FFN_H, 0, 3);   // global_prefetch_b8
        va = *(const float4*)p;
        vb = *(const float4*)(p + FFN_H);
      }
      acch = __builtin_amdgcn_wmma_f32_16x16x32_bf16(
          false, a, false, b, (short)0, acch, false, false);
      if (hasNext) stw1((i + 1) & 1, va, vb);
    }
    // bias + exact gelu, store h as bf16
    {
      float bv = b1[e * FFN_H + h0 + ni * 16 + ln16];
      int nl = ni * 16 + ln16;
#pragma unroll
      for (int j = 0; j < 8; ++j) {
        float v = acch[j] + bv;
        float g = 0.5f * v * (1.0f + erff(v * 0.70710678118654752f));
        h_lds[mi * 16 + j + lhi * 8][nl] = (__bf16)g;
      }
    }
    __syncthreads();  // h visible to all waves

    // ---------------- stage B: y += h @ W2[h0:h0+64, :]  (wave owns 128 cols)
    // software pipeline over s = (kk2-half, nj), per-wave double-buffered LDS
    float4 wv[4];
    ldw2(0, h0, wv);
    stw2(0, wv);
    v16bf a2[2];
#pragma unroll
    for (int s = 0; s < 16; ++s) {
      if ((s & 7) == 0) {
        int kk2 = (s >> 3) * 32;
#pragma unroll
        for (int m2 = 0; m2 < 2; ++m2) {
          int ml = m2 * 16 + ln16;
          int ks = lhi * 8;
          a2[m2] = frag16(&h_lds[ml][kk2 + ks], &h_lds[ml][kk2 + ks + 16]);
        }
      }
      v16bf bfr =
          frag16(&w2t[w][s & 1][ln16][lhi * 8], &w2t[w][s & 1][ln16][lhi * 8 + 16]);
      if (s < 15) ldw2(s + 1, h0, wv);           // next slab in flight
      accy[0][s & 7] = __builtin_amdgcn_wmma_f32_16x16x32_bf16(
          false, a2[0], false, bfr, (short)0, accy[0][s & 7], false, false);
      accy[1][s & 7] = __builtin_amdgcn_wmma_f32_16x16x32_bf16(
          false, a2[1], false, bfr, (short)0, accy[1][s & 7], false, false);
      if (s < 15) stw2((s + 1) & 1, wv);         // after WMMAs: no ds stall
    }
    // next chunk's stage-A barriers protect h_lds / w1t reuse
  }

  // ---------------- epilogue: +b2, scale by gate weight, atomic combine
#pragma unroll
  for (int m2 = 0; m2 < 2; ++m2) {
#pragma unroll
    for (int nj = 0; nj < 8; ++nj) {
      int ncol = w * 128 + nj * 16 + ln16;
      float bb = b2[e * D_MODEL + ncol];
#pragma unroll
      for (int j = 0; j < 8; ++j) {
        int ml = m2 * 16 + j + lhi * 8;
        float val = (accy[m2][nj][j] + bb) * s_wgt[ml];
        atomicAdd(&out[(size_t)s_tok[ml] * D_MODEL + ncol], val);
      }
    }
  }
}

// ---------------------------------------------------------------- launcher
extern "C" void kernel_launch(void* const* d_in, const int* in_sizes, int n_in,
                              void* d_out, int out_size, void* d_ws, size_t ws_size,
                              hipStream_t stream) {
  const float* x  = (const float*)d_in[0];
  const float* wg = (const float*)d_in[1];
  const float* W1 = (const float*)d_in[2];
  const float* b1 = (const float*)d_in[3];
  const float* W2 = (const float*)d_in[4];
  const float* b2 = (const float*)d_in[5];
  float* out = (float*)d_out;

  int*   cnt = (int*)d_ws;
  int*   tok = cnt + 16;                        // 8*CAP ints
  float* wgt = (float*)(tok + N_EXP * CAP);     // 8*CAP floats

  moe_zero_kernel<<<(N_TOK * D_MODEL + 255) / 256, 256, 0, stream>>>(out, cnt);
  moe_gate_kernel<<<N_TOK / 256, 256, 0, stream>>>(x, wg, cnt, tok, wgt);

  dim3 grid(CAP / BM, N_EXP);   // 256 tiles x 8 experts; blocks past count exit
  moe_ffn_kernel<<<grid, 256, 0, stream>>>(x, W1, b1, W2, b2,
                                           cnt, tok, wgt, out);
}
